// nonLocal_69028714382065
// MI455X (gfx1250) — compile-verified, matrix operand order checked by reading, not run on hardware
//
#include <hip/hip_runtime.h>
#include <hip/hip_bf16.h>

// Pipeline: strided-conv-as-GEMM -> 3x 1x1-proj GEMM -> fused non-local
// attention (sigmoid in-LDS, flash style) -> bilinear x8 upsample.
// GEMMs use v_wmma_f32_16x16x32_f16; f16 tile staging uses
// global_load_async_to_lds_b128 (ASYNCcnt) where no conversion is needed.

typedef __attribute__((ext_vector_type(16))) _Float16 v16h;
typedef __attribute__((ext_vector_type(8)))  _Float16 v8h;
typedef __attribute__((ext_vector_type(8)))  float    v8f;
typedef __attribute__((ext_vector_type(4)))  float    f4;

__device__ __forceinline__ v8h cvt8(const float* p) {
  f4 a = *(const f4*)p;
  f4 b = *(const f4*)(p + 4);
  v8h r;
#pragma unroll
  for (int e = 0; e < 4; ++e) { r[e] = (_Float16)a[e]; r[e + 4] = (_Float16)b[e]; }
  return r;
}

// Async 16-byte global -> LDS copy (CDNA5, tracked by ASYNCcnt).
// Generic pointers to __shared__ carry the LDS byte offset in the low 32 bits
// (aperture scheme, ISA 10.2), so truncation yields the VDST LDS address.
__device__ __forceinline__ void async_g2l_b128(void* lds, const void* gptr) {
  unsigned l = (unsigned)(uintptr_t)lds;
  unsigned long long g = (unsigned long long)(uintptr_t)gptr;
  asm volatile("global_load_async_to_lds_b128 %0, %1, off"
               :: "v"(l), "v"(g) : "memory");
}
__device__ __forceinline__ void wait_async() {
  asm volatile("s_wait_asynccnt 0x0" ::: "memory");
}

// Load a 16(row) x 32(K) f16 WMMA operand fragment from an LDS tile stored
// row-major with `stride` f16 per row (rows = M for A, rows = N for B).
// Lane layout per CDNA5 ISA 7.12.2: lanes 0-15 hold K {0..7,16..23},
// lanes 16-31 hold K {8..15,24..31} of the same row.
__device__ __forceinline__ v16h load_frag(const _Float16* lds, int row0,
                                          int stride, int k0) {
  const int lane = threadIdx.x & 31;
  const _Float16* p = lds + (row0 + (lane & 15)) * stride + k0 + ((lane >> 4) << 3);
  v8h lo = *(const v8h*)p;         // K + {0..7} / {8..15}
  v8h hi = *(const v8h*)(p + 16);  // K + {16..23} / {24..31}
  v16h r;
#pragma unroll
  for (int t = 0; t < 8; ++t) { r[t] = lo[t]; r[t + 8] = hi[t]; }
  return r;
}

#define WMMA_F16(a, b, c) \
  __builtin_amdgcn_wmma_f32_16x16x32_f16(false, (a), false, (b), (short)0, (c), false, false)

// ---------------------------------------------------------------------------
// Kernel 1: 8x8 stride-8 conv == GEMM  [256 oc] x [8192 pix] x [K=4096]
// BN+ReLU epilogue; y stored f16 pixel-major [8192][256].
// (Staging keeps the VGPR path: fp32 -> f16 conversion happens in flight.)
// ---------------------------------------------------------------------------
__global__ __launch_bounds__(128) void conv8x8_wmma(
    const float* __restrict__ x, const float* __restrict__ w1,
    const float* __restrict__ s1, const float* __restrict__ b1,
    _Float16* __restrict__ y16) {
  __shared__ _Float16 As[64 * 32];
  __shared__ _Float16 Bs[64 * 32];
  const int tid = threadIdx.x, lane = tid & 31, wid = tid >> 5;
  const int n0 = blockIdx.x * 64;  // pixel tile
  const int m0 = blockIdx.y * 64;  // out-channel tile
  const int wm = (wid >> 1) * 32, wn = (wid & 1) * 32;

  v8f acc[2][2] = {};
  const int ml = tid >> 1, half = tid & 1;  // loader: 2 threads per tile row
  const int pg = n0 + ml;                   // this loader thread's pixel
  const int pb = pg >> 10, pp = pg & 1023;
  const int ph = pp >> 5, pw = pp & 31;

  for (int kk = 0; kk < 4096; kk += 32) {
    // A tile: w1 is [256][4096] contiguous (oc)(ic*64+kh*8+kw)
#pragma unroll
    for (int g = 0; g < 2; ++g) {
      const int k = kk + half * 16 + g * 8;
      *(v8h*)&As[ml * 32 + half * 16 + g * 8] = cvt8(w1 + (size_t)(m0 + ml) * 4096 + k);
    }
    // B tile: im2col of x[b, ic, ph*8+kh, pw*8+kw]; 8 consecutive kw = 8 floats
#pragma unroll
    for (int g = 0; g < 2; ++g) {
      const int k = kk + half * 16 + g * 8;
      const int ic = k >> 6, rem = k & 63, kh = rem >> 3;
      const float* src =
          x + ((size_t)(pb * 64 + ic) * 256 + (ph * 8 + kh)) * 256 + pw * 8;
      *(v8h*)&Bs[ml * 32 + half * 16 + g * 8] = cvt8(src);
    }
    if (kk + 32 < 4096)
      __builtin_prefetch(w1 + (size_t)(m0 + ml) * 4096 + kk + 32, 0, 1);
    __syncthreads();
    v16h a0 = load_frag(As, wm, 32, 0);
    v16h a1 = load_frag(As, wm + 16, 32, 0);
    v16h t0 = load_frag(Bs, wn, 32, 0);
    v16h t1 = load_frag(Bs, wn + 16, 32, 0);
    acc[0][0] = WMMA_F16(a0, t0, acc[0][0]);
    acc[0][1] = WMMA_F16(a0, t1, acc[0][1]);
    acc[1][0] = WMMA_F16(a1, t0, acc[1][0]);
    acc[1][1] = WMMA_F16(a1, t1, acc[1][1]);
    __syncthreads();
  }
  const int hi8 = (lane >> 4) << 3, nlo = lane & 15;
#pragma unroll
  for (int tm = 0; tm < 2; ++tm)
#pragma unroll
    for (int tn = 0; tn < 2; ++tn)
#pragma unroll
      for (int r = 0; r < 8; ++r) {
        const int oc = m0 + wm + tm * 16 + r + hi8;
        const int pix = n0 + wn + tn * 16 + nlo;
        const float v = acc[tm][tn][r] * s1[oc] + b1[oc];
        y16[(size_t)pix * 256 + oc] = (_Float16)fmaxf(v, 0.f);
      }
}

// ---------------------------------------------------------------------------
// Kernel 2: 1x1 projection  [128 o] x [8192 pix] x [K=256] + BN + ReLU.
// Output f16 pixel-major [pix][128]. B tiles staged via async-to-LDS.
// ---------------------------------------------------------------------------
__global__ __launch_bounds__(128) void proj_wmma(
    const _Float16* __restrict__ y16, const float* __restrict__ W,
    const float* __restrict__ sc, const float* __restrict__ bi,
    _Float16* __restrict__ out16) {
  __shared__ _Float16 As[128 * 32];
  __shared__ _Float16 Bs[64 * 32];
  const int tid = threadIdx.x, lane = tid & 31, wid = tid >> 5;
  const int n0 = blockIdx.x * 64;
  const int m0w = wid * 32;
  v8f acc[2][4] = {};
  const int nl = tid >> 1, half = tid & 1;
  for (int kk = 0; kk < 256; kk += 32) {
    // B: y16 is pixel-major, contiguous along K=c -> pure async 16B copies
    async_g2l_b128(&Bs[nl * 32 + half * 16],
                   y16 + (size_t)(n0 + nl) * 256 + kk + half * 16);
    async_g2l_b128(&Bs[nl * 32 + half * 16 + 8],
                   y16 + (size_t)(n0 + nl) * 256 + kk + half * 16 + 8);
    // A: W[128][256] fp32 -> f16 K-chunk (one row per thread, converts in VGPRs)
#pragma unroll
    for (int g = 0; g < 4; ++g)
      *(v8h*)&As[tid * 32 + g * 8] = cvt8(W + (size_t)tid * 256 + kk + g * 8);
    wait_async();
    __syncthreads();
    v16h a0 = load_frag(As, m0w, 32, 0);
    v16h a1 = load_frag(As, m0w + 16, 32, 0);
#pragma unroll
    for (int tn = 0; tn < 4; ++tn) {
      v16h bf = load_frag(Bs, tn * 16, 32, 0);
      acc[0][tn] = WMMA_F16(a0, bf, acc[0][tn]);
      acc[1][tn] = WMMA_F16(a1, bf, acc[1][tn]);
    }
    __syncthreads();
  }
  const int hi8 = (lane >> 4) << 3, nlo = lane & 15;
#pragma unroll
  for (int tm = 0; tm < 2; ++tm)
#pragma unroll
    for (int tn = 0; tn < 4; ++tn)
#pragma unroll
      for (int r = 0; r < 8; ++r) {
        const int oc = m0w + tm * 16 + r + hi8;
        const int pix = n0 + tn * 16 + nlo;
        const float v = acc[tm][tn][r] * sc[oc] + bi[oc];
        out16[(size_t)pix * 128 + oc] = (_Float16)fmaxf(v, 0.f);
      }
}

// ---------------------------------------------------------------------------
// Kernel 3: fused non-local attention, per (batch, 64-row i-tile).
//   S_ij = sum_c fai[c,i] sit[c,j]  -> sigmoid -> f16 S-tile in LDS
//   out[i,o] += S * gama[o,j]       (accumulated in registers over j chunks)
// fai/sit tiles staged via async-to-LDS; the 1024x1024 S never hits global.
// ---------------------------------------------------------------------------
__global__ __launch_bounds__(128) void attn_wmma(
    const _Float16* __restrict__ fai, const _Float16* __restrict__ sit,
    const _Float16* __restrict__ gam, float* __restrict__ att) {
  __shared__ _Float16 Af[64 * 128];  // fai  [i][c]
  __shared__ _Float16 Bs[64 * 128];  // sit  [j][c]
  __shared__ _Float16 Gs[128 * 64];  // gama [o][j]  (transposed on load)
  __shared__ _Float16 Ss[64 * 64];   // sigmoid(S) [i][j]
  const int tid = threadIdx.x, lane = tid & 31, wid = tid >> 5;
  const int b = blockIdx.y;
  const int i0 = blockIdx.x * 64;
  const size_t pbase = (size_t)b * 1024;
  const int hi8 = (lane >> 4) << 3, nlo = lane & 15;

  for (int idx = tid; idx < 1024; idx += 128) {  // fai i-tile, async
    const int row = idx >> 4, c8 = (idx & 15) << 3;
    async_g2l_b128(&Af[row * 128 + c8], fai + (pbase + i0 + row) * 128 + c8);
  }

  v8f oacc[8] = {};
  for (int j0 = 0; j0 < 1024; j0 += 64) {
    for (int idx = tid; idx < 1024; idx += 128) {  // sit chunk [j][c], async
      const int row = idx >> 4, c8 = (idx & 15) << 3;
      async_g2l_b128(&Bs[row * 128 + c8], sit + (pbase + j0 + row) * 128 + c8);
    }
    {  // gama chunk, transposed into [o][j] (element scatter -> VGPR path)
      const int jl = tid >> 1, oh = (tid & 1) * 64;
      const _Float16* src = gam + (pbase + j0 + jl) * 128 + oh;
#pragma unroll
      for (int v = 0; v < 8; ++v) {
        v8h g = *(const v8h*)(src + v * 8);
#pragma unroll
        for (int e = 0; e < 8; ++e) Gs[(oh + v * 8 + e) * 64 + jl] = g[e];
      }
    }
    wait_async();
    __syncthreads();
    // GEMM1: 16i x 64j per wave, K = 128 channels
    v8f sacc[4] = {};
#pragma unroll
    for (int ks = 0; ks < 4; ++ks) {
      v16h af = load_frag(Af, wid * 16, 128, ks * 32);
#pragma unroll
      for (int jt = 0; jt < 4; ++jt) {
        v16h bf = load_frag(Bs, jt * 16, 128, ks * 32);
        sacc[jt] = WMMA_F16(af, bf, sacc[jt]);
      }
    }
    // sigmoid, write S-tile (this is also the C-layout -> A-layout transpose)
#pragma unroll
    for (int jt = 0; jt < 4; ++jt)
#pragma unroll
      for (int r = 0; r < 8; ++r) {
        const int il = wid * 16 + r + hi8;
        const int jl = jt * 16 + nlo;
        const float v = sacc[jt][r];
        Ss[il * 64 + jl] = (_Float16)(1.0f / (1.0f + __expf(-v)));
      }
    __syncthreads();
    // GEMM2: 16i x 128o per wave, K = 64 j-values of this chunk
#pragma unroll
    for (int ks = 0; ks < 2; ++ks) {
      v16h aS = load_frag(Ss, wid * 16, 64, ks * 32);
#pragma unroll
      for (int ot = 0; ot < 8; ++ot) {
        v16h bG = load_frag(Gs, ot * 16, 64, ks * 32);
        oacc[ot] = WMMA_F16(aS, bG, oacc[ot]);
      }
    }
    __syncthreads();
  }
#pragma unroll
  for (int ot = 0; ot < 8; ++ot)
#pragma unroll
    for (int r = 0; r < 8; ++r) {
      const int o = ot * 16 + nlo;
      const int i = i0 + wid * 16 + r + hi8;
      att[((size_t)b * 128 + o) * 1024 + i] = oacc[ot][r];
    }
}

// ---------------------------------------------------------------------------
// Kernel 4: bilinear x8 upsample, align_corners=True. Pure bandwidth.
// ---------------------------------------------------------------------------
__global__ __launch_bounds__(256) void upsample8_ac(
    const float* __restrict__ att, float* __restrict__ out) {
  const unsigned g = blockIdx.x * 256u + threadIdx.x;  // 2^24 threads total
  const int ox0 = (g & 63) << 2;
  const int oy = (g >> 6) & 255;
  const int c = (g >> 14) & 127;
  const int b = g >> 21;
  const float scl = 31.0f / 255.0f;
  const float fy = oy * scl;
  const int y0 = (int)fy;
  const float wy = fy - (float)y0;
  const int y1 = min(y0 + 1, 31);
  const float* base = att + ((size_t)b * 128 + c) * 1024;
  f4 res;
#pragma unroll
  for (int t = 0; t < 4; ++t) {
    const int ox = ox0 + t;
    const float fx = ox * scl;
    const int x0 = (int)fx;
    const float wx = fx - (float)x0;
    const int x1 = min(x0 + 1, 31);
    const float a00 = base[y0 * 32 + x0], a01 = base[y0 * 32 + x1];
    const float a10 = base[y1 * 32 + x0], a11 = base[y1 * 32 + x1];
    const float top = a00 * (1.0f - wx) + a01 * wx;
    const float bot = a10 * (1.0f - wx) + a11 * wx;
    res[t] = top * (1.0f - wy) + bot * wy;
  }
  *(f4*)(out + (size_t)g * 4) = res;
}

extern "C" void kernel_launch(void* const* d_in, const int* in_sizes, int n_in,
                              void* d_out, int out_size, void* d_ws, size_t ws_size,
                              hipStream_t stream) {
  const float* x    = (const float*)d_in[0];
  const float* w1   = (const float*)d_in[1];
  const float* bn1s = (const float*)d_in[2];
  const float* bn1b = (const float*)d_in[3];
  const float* wf   = (const float*)d_in[4];
  const float* sf   = (const float*)d_in[5];
  const float* bf   = (const float*)d_in[6];
  const float* wst  = (const float*)d_in[7];
  const float* ssc  = (const float*)d_in[8];
  const float* bsb  = (const float*)d_in[9];
  const float* wg   = (const float*)d_in[10];
  const float* sg   = (const float*)d_in[11];
  const float* bg   = (const float*)d_in[12];

  char* ws = (char*)d_ws;                                // workspace map (14 MB)
  _Float16* y16   = (_Float16*)(ws);                     // [8192][256] f16, 4 MB
  _Float16* fai16 = (_Float16*)(ws + (4u  << 20));       // [8192][128] f16, 2 MB
  _Float16* sit16 = (_Float16*)(ws + (6u  << 20));       // [8192][128] f16, 2 MB
  _Float16* gam16 = (_Float16*)(ws + (8u  << 20));       // [8192][128] f16, 2 MB
  float*    att   = (float*)  (ws + (10u << 20));        // [8][128][1024] f32, 4 MB
  float* out = (float*)d_out;

  conv8x8_wmma<<<dim3(128, 4), 128, 0, stream>>>(x, w1, bn1s, bn1b, y16);
  proj_wmma<<<dim3(128), 128, 0, stream>>>(y16, wf, sf, bf, fai16);
  proj_wmma<<<dim3(128), 128, 0, stream>>>(y16, wst, ssc, bsb, sit16);
  proj_wmma<<<dim3(128), 128, 0, stream>>>(y16, wg, sg, bg, gam16);
  attn_wmma<<<dim3(16, 8), 128, 0, stream>>>(fai16, sit16, gam16, att);
  upsample8_ac<<<dim3(65536), 256, 0, stream>>>(att, out);
}
